// RecurrentLayer_45157286150494
// MI455X (gfx1250) — matrix operand order repaired and checked by reading, not compiled, not run on hardware
//
#include <hip/hip_runtime.h>
#include <hip/hip_bf16.h>

// Problem constants (match reference).
#define B_ 64
#define T_ 1024
#define I_ 512
#define H_ 1024

typedef __attribute__((ext_vector_type(16))) __bf16 v16bf;
typedef __attribute__((ext_vector_type(8)))  __bf16 v8bf;
typedef __attribute__((ext_vector_type(8)))  float  v8f;

union AFrag { v16bf v; v8bf h[2]; };

static __device__ __forceinline__ v8f wmma_bf16(const v16bf& a, const v16bf& b, v8f c) {
  return __builtin_amdgcn_wmma_f32_16x16x32_bf16(false, a, false, b, (short)0, c,
                                                 false, false);
}

// ---------------------------------------------------------------------------
// Split fp32 W[K][N] into bf16 hi/lo, pre-swizzled into WMMA B-fragment order:
// blocks of 32(K) x 16(N); within a block, half index = lane*16 + j where
// lane in [0,32), j in [0,16): element W[kt*32 + (j>>3)*16 + (lane>>4)*8 + (j&7)][nt*16 + (lane&15)].
// So a wave's B-fragment = one v16bf (32B) per lane, fully contiguous.
// ---------------------------------------------------------------------------
__global__ void split_swizzle_kernel(const float* __restrict__ W,
                                     __bf16* __restrict__ hi,
                                     __bf16* __restrict__ lo,
                                     int K, int N) {
  int e = blockIdx.x * blockDim.x + threadIdx.x;
  int total = K * N;
  if (e >= total) return;
  int NT = N >> 4;
  int block  = e >> 9;     // 512 halves per 32x16 block
  int within = e & 511;
  int l = within >> 4;
  int j = within & 15;
  int kt = block / NT;
  int nt = block - kt * NT;
  int k = kt * 32 + ((j >> 3) << 4) + ((l >> 4) << 3) + (j & 7);
  int n = nt * 16 + (l & 15);
  float w = W[(size_t)k * N + n];
  __bf16 h = (__bf16)w;
  hi[e] = h;
  lo[e] = (__bf16)(w - (float)h);
}

// ---------------------------------------------------------------------------
// Kernel 1: xp = x @ W_in + b, written into d_out (consumed in-place by scan).
// Grid: (M/128, H/64); 256 threads = 8 waves. Wave w computes a 16x64 strip
// (rows w*16..w*16+15 of the 128-row tile, all 64 cols of the WG's N-slice).
// x tile is staged in LDS as bf16 hi/lo per 32-deep K chunk. bf16x3 products.
// ---------------------------------------------------------------------------
__global__ __launch_bounds__(256) void xproj_wmma_kernel(
    const float* __restrict__ x,        // [M, I]
    const v16bf* __restrict__ whi,      // swizzled [I/32][H/16] blocks
    const v16bf* __restrict__ wlo,
    const float* __restrict__ bias,     // [H]
    float* __restrict__ out) {          // [M, H]
  __shared__ __bf16 xhi[128 * 32];
  __shared__ __bf16 xlo[128 * 32];

  const int tid  = threadIdx.x;
  const int w    = tid >> 5;
  const int l    = tid & 31;
  const int half = l >> 4;
  const int lm   = l & 15;
  const int gm0  = blockIdx.x * 128;
  const int n0   = blockIdx.y * 64;

  const v8f vzero = {0.f, 0.f, 0.f, 0.f, 0.f, 0.f, 0.f, 0.f};
  v8f acc[4] = {vzero, vzero, vzero, vzero};

  for (int kt = 0; kt < I_ / 32; ++kt) {
    __syncthreads();  // previous chunk's LDS reads done
    for (int i = tid; i < 128 * 32; i += 256) {
      int m  = i >> 5;
      int kk = i & 31;
      float v = x[(size_t)(gm0 + m) * I_ + kt * 32 + kk];
      __bf16 h = (__bf16)v;
      xhi[i] = h;
      xlo[i] = (__bf16)(v - (float)h);
    }
    __syncthreads();

    AFrag ahi, alo;
    const int mrow = w * 16 + lm;
    ahi.h[0] = *(const v8bf*)&xhi[mrow * 32 + half * 8];
    ahi.h[1] = *(const v8bf*)&xhi[mrow * 32 + 16 + half * 8];
    alo.h[0] = *(const v8bf*)&xlo[mrow * 32 + half * 8];
    alo.h[1] = *(const v8bf*)&xlo[mrow * 32 + 16 + half * 8];

#pragma unroll
    for (int j = 0; j < 4; ++j) {
      size_t blk = ((size_t)kt * (H_ / 16) + (blockIdx.y * 4 + j)) * 32 + l;
      v16bf bh = whi[blk];
      v16bf bl = wlo[blk];
      acc[j] = wmma_bf16(ahi.v, bh, acc[j]);  // hi*hi
      acc[j] = wmma_bf16(ahi.v, bl, acc[j]);  // hi*lo
      acc[j] = wmma_bf16(alo.v, bh, acc[j]);  // lo*hi
    }
  }

#pragma unroll
  for (int j = 0; j < 4; ++j) {
    int n = n0 + j * 16 + lm;
    float bv = bias[n];
#pragma unroll
    for (int r = 0; r < 8; ++r) {
      int m = r + half * 8;  // C/D layout: VGPR r -> row r (+8 for upper half-wave)
      size_t row = (size_t)gm0 + w * 16 + m;
      out[row * H_ + n] = acc[j][r] + bv;
    }
  }
}

// ---------------------------------------------------------------------------
// Kernel 2: persistent recurrence scan. 4 workgroups x 512 threads (16 waves).
// Each WG owns 16 batch rows; h[16,1024] lives in LDS as bf16 hi/lo (64 KB).
// Wave w owns 64 output columns (4 WMMA N-tiles). Per step: 32 K-chunks x
// 4 N-tiles x 3 bf16 WMMAs; W_h streams from L2 in fragment-linear blocks.
// d_out holds xp on entry and is overwritten in place with h_t.
// Only intra-workgroup barriers -> no global sync on the serial chain.
// ---------------------------------------------------------------------------
__global__ __launch_bounds__(512) void rnn_scan_wmma_kernel(
    const v16bf* __restrict__ whi,   // W_h swizzled [H/32][H/16] blocks
    const v16bf* __restrict__ wlo,
    float* __restrict__ out) {       // [B, T, H]
  __shared__ __bf16 hhi[16 * H_];
  __shared__ __bf16 hlo[16 * H_];

  const int tid  = threadIdx.x;
  const int w    = tid >> 5;
  const int l    = tid & 31;
  const int half = l >> 4;
  const int lm   = l & 15;
  const int b0   = blockIdx.x * 16;

  for (int i = tid; i < 16 * H_; i += 512) {
    hhi[i] = (__bf16)0.0f;
    hlo[i] = (__bf16)0.0f;
  }
  __syncthreads();

  const v8f vzero = {0.f, 0.f, 0.f, 0.f, 0.f, 0.f, 0.f, 0.f};
  float y[4][8];

  for (int t = 0; t < T_; ++t) {
    v8f acc[4] = {vzero, vzero, vzero, vzero};

    for (int kt = 0; kt < H_ / 32; ++kt) {
      AFrag ahi, alo;
      const int kb = kt * 32;
      ahi.h[0] = *(const v8bf*)&hhi[lm * H_ + kb + half * 8];
      ahi.h[1] = *(const v8bf*)&hhi[lm * H_ + kb + 16 + half * 8];
      alo.h[0] = *(const v8bf*)&hlo[lm * H_ + kb + half * 8];
      alo.h[1] = *(const v8bf*)&hlo[lm * H_ + kb + 16 + half * 8];
#pragma unroll
      for (int j = 0; j < 4; ++j) {
        size_t blk = ((size_t)kt * (H_ / 16) + (w * 4 + j)) * 32 + l;
        v16bf bh = whi[blk];
        v16bf bl = wlo[blk];
        acc[j] = wmma_bf16(ahi.v, bh, acc[j]);
        acc[j] = wmma_bf16(ahi.v, bl, acc[j]);
        acc[j] = wmma_bf16(alo.v, bh, acc[j]);
      }
    }

    // h_t = tanh(acc + xp); write to out, keep for LDS update.
#pragma unroll
    for (int j = 0; j < 4; ++j) {
      int n = w * 64 + j * 16 + lm;
#pragma unroll
      for (int r = 0; r < 8; ++r) {
        int m = r + half * 8;
        size_t off = ((size_t)(b0 + m) * T_ + t) * H_ + n;
        float v = tanhf(acc[j][r] + out[off]);
        out[off] = v;
        y[j][r] = v;
      }
    }

    __syncthreads();  // all waves finished reading h_{t-1}
#pragma unroll
    for (int j = 0; j < 4; ++j) {
      int n = w * 64 + j * 16 + lm;
#pragma unroll
      for (int r = 0; r < 8; ++r) {
        int m = r + half * 8;
        float v = y[j][r];
        __bf16 hv = (__bf16)v;
        hhi[m * H_ + n] = hv;
        hlo[m * H_ + n] = (__bf16)(v - (float)hv);
      }
    }
    __syncthreads();  // h_t visible to every wave
  }
}

// ---------------------------------------------------------------------------
extern "C" void kernel_launch(void* const* d_in, const int* in_sizes, int n_in,
                              void* d_out, int out_size, void* d_ws, size_t ws_size,
                              hipStream_t stream) {
  (void)in_sizes; (void)n_in; (void)out_size; (void)ws_size;
  const float* x    = (const float*)d_in[0];  // [B,T,I]
  const float* wh   = (const float*)d_in[1];  // [H,H]
  const float* wi   = (const float*)d_in[2];  // [I,H]
  const float* bias = (const float*)d_in[3];  // [H]
  float* out = (float*)d_out;                 // [B,T,H]

  // Workspace layout (6 MB total): bf16 hi/lo splits of W_in and W_h.
  char* ws = (char*)d_ws;
  __bf16* wi_hi = (__bf16*)(ws);
  __bf16* wi_lo = (__bf16*)(ws + (size_t)I_ * H_ * 2);
  __bf16* wh_hi = (__bf16*)(ws + (size_t)I_ * H_ * 4);
  __bf16* wh_lo = (__bf16*)(ws + (size_t)I_ * H_ * 4 + (size_t)H_ * H_ * 2);

  {
    int total = I_ * H_;
    split_swizzle_kernel<<<(total + 255) / 256, 256, 0, stream>>>(wi, wi_hi, wi_lo, I_, H_);
  }
  {
    int total = H_ * H_;
    split_swizzle_kernel<<<(total + 255) / 256, 256, 0, stream>>>(wh, wh_hi, wh_lo, H_, H_);
  }

  dim3 g1((B_ * T_) / 128, H_ / 64);
  xproj_wmma_kernel<<<g1, 256, 0, stream>>>(x, (const v16bf*)wi_hi, (const v16bf*)wi_lo,
                                            bias, out);

  rnn_scan_wmma_kernel<<<B_ / 16, 512, 0, stream>>>((const v16bf*)wh_hi,
                                                    (const v16bf*)wh_lo, out);
}
